// SelfAttention_39281770889545
// MI455X (gfx1250) — matrix operand order repaired and checked by reading, not compiled, not run on hardware
//
#include <hip/hip_runtime.h>

// ---------------- problem constants ----------------
#define NB      4        // batch
#define TSEQ    8192     // sequence length
#define DIMX    1024     // model dim (== K == N of all GEMMs)
#define DH      64       // head dim
#define LHEADS  4        // local-attn heads
#define WIN     128      // local attention window
#define BSPOT   64       // linear-attn chunk

typedef __attribute__((ext_vector_type(16))) __bf16 v16bf;
typedef __attribute__((ext_vector_type(8)))  float  v8f;
typedef int v4i __attribute__((vector_size(16)));   // matches builtin's param type

union Frag { v16bf v; uint4 q[2]; unsigned int u[8]; };

// gfx1250 async global->LDS path (ASYNCcnt), guarded so either toolchain compiles
#if defined(__HIP_DEVICE_COMPILE__) && \
    __has_builtin(__builtin_amdgcn_global_load_async_to_lds_b128) && \
    __has_builtin(__builtin_amdgcn_s_wait_asynccnt)
#define HAVE_ASYNC_LDS 1
#else
#define HAVE_ASYNC_LDS 0
#endif

__device__ __forceinline__ unsigned short f2bf(float f) {
    unsigned int u = __float_as_uint(f);
    unsigned int r = u + 0x7FFFu + ((u >> 16) & 1u);   // round-to-nearest-even
    return (unsigned short)(r >> 16);
}
__device__ __forceinline__ float bf2f(unsigned short h) {
    return __uint_as_float(((unsigned int)h) << 16);
}

// ---------------- prep: f32 -> bf16 ----------------
__global__ void cvt_bf16_kernel(const float* __restrict__ in,
                                unsigned short* __restrict__ out, int n) {
    int i = blockIdx.x * 256 + threadIdx.x;
    if (i < n) out[i] = f2bf(in[i]);
}

// W[k][n] (f32, 1024x1024) -> Wt[n][k] (bf16) so B-fragments load contiguous K
__global__ void transpose_w_kernel(const float* __restrict__ w,
                                   unsigned short* __restrict__ wt) {
    int i = blockIdx.x * 256 + threadIdx.x;       // 1M elements
    int n = i >> 10, k = i & 1023;
    wt[(size_t)n * DIMX + k] = f2bf(w[(size_t)k * DIMX + n]);
}

// Stage a 16-row x 1024-col bf16 A-panel (32KB) into LDS, all 256 threads.
__device__ __forceinline__ void stage_A_panel(const unsigned short* __restrict__ src,
                                              int m0, uint4* sA) {
    for (int i = threadIdx.x; i < 2048; i += 256) {          // 2048 x 16B chunks
        const unsigned short* g = src + (size_t)(m0 + (i >> 7)) * DIMX + ((i & 127) << 3);
#if HAVE_ASYNC_LDS
        __builtin_amdgcn_global_load_async_to_lds_b128(
            (__attribute__((address_space(1))) v4i*)(v4i*)g,
            (__attribute__((address_space(3))) v4i*)(v4i*)&sA[i],
            0, 0);
#else
        sA[i] = *(const uint4*)g;
#endif
    }
#if HAVE_ASYNC_LDS
    __builtin_amdgcn_s_wait_asynccnt(0);
#endif
    __syncthreads();
}

// Load A fragment (16x32 MxK, ISA 7.12.2 16-bit layout) from the LDS panel.
__device__ __forceinline__ void load_A_frag(Frag& a, const char* sAc,
                                            int lid, int hl, int k0) {
    const int base = lid * 2048 + (k0 << 1) + (hl << 4);
    a.q[0] = *(const uint4*)(sAc + base);
    a.q[1] = *(const uint4*)(sAc + base + 32);
}

// Load B fragment (32x16 KxN): lane = column; two 16B K runs at k0+8*hl and k0+16+8*hl.
__device__ __forceinline__ void load_B_frag(Frag& f, const unsigned short* wt,
                                            size_t colB, int hl, int k0) {
    const unsigned short* p = wt + colB + k0 + (hl << 3);
    f.q[0] = *(const uint4*)p;
    f.q[1] = *(const uint4*)(p + 16);
}

// ---------------- fused QKV GEMM ----------------
// Block = 256 threads (8 waves). Block tile: 16 rows x 128 cols for each of q,k,v.
// A panel staged once in LDS (async-to-LDS on gfx1250); 3 WMMA per wave per k-step.
__global__ __launch_bounds__(256)
void qkv_gemm_kernel(const unsigned short* __restrict__ xb,
                     const unsigned short* __restrict__ wq,
                     const unsigned short* __restrict__ wk,
                     const unsigned short* __restrict__ wv,
                     unsigned short* __restrict__ qb,
                     unsigned short* __restrict__ kb,
                     unsigned short* __restrict__ vb) {
    __shared__ uint4 sA[2048];                    // 16 x 1024 bf16 = 32KB

    const int wave = threadIdx.x >> 5;
    const int lane = threadIdx.x & 31;
    const int mt = blockIdx.x >> 3;               // 0..2047
    const int ng = blockIdx.x & 7;                // 0..7 N-groups of 128
    const int m0 = mt * 16;
    const int n0 = ng * 128 + wave * 16;
    const int hl  = lane >> 4;
    const int lid = lane & 15;

    stage_A_panel(xb, m0, sA);
    const char* sAc = (const char*)sA;

    v8f cq = {}; v8f ck = {}; v8f cv = {};
    const size_t colB = (size_t)(n0 + lid) * DIMX;

    for (int k0 = 0; k0 < DIMX; k0 += 32) {
        __builtin_prefetch(wq + colB + k0 + 32, 0, 0);   // global_prefetch_b8
        Frag a, fq, fk, fv;
        load_A_frag(a, sAc, lid, hl, k0);
        load_B_frag(fq, wq, colB, hl, k0);
        load_B_frag(fk, wk, colB, hl, k0);
        load_B_frag(fv, wv, colB, hl, k0);
        cq = __builtin_amdgcn_wmma_f32_16x16x32_bf16(false, a.v, false, fq.v,
                                                     (short)0, cq, false, false);
        ck = __builtin_amdgcn_wmma_f32_16x16x32_bf16(false, a.v, false, fk.v,
                                                     (short)0, ck, false, false);
        cv = __builtin_amdgcn_wmma_f32_16x16x32_bf16(false, a.v, false, fv.v,
                                                     (short)0, cv, false, false);
    }
    #pragma unroll
    for (int g = 0; g < 8; ++g) {                 // ISA C/D layout
        int row = m0 + g + 8 * hl;
        int col = n0 + lid;
        size_t off = (size_t)row * DIMX + col;
        qb[off] = f2bf(cq[g]);
        kb[off] = f2bf(ck[g]);
        vb[off] = f2bf(cv[g]);
    }
}

// ---------------- local windowed causal attention (heads 0..3) ---------------
__global__ __launch_bounds__(128)
void local_attn_kernel(const unsigned short* __restrict__ qb,
                       const unsigned short* __restrict__ kb,
                       const unsigned short* __restrict__ vb,
                       unsigned short* __restrict__ attnb) {
    const int bucket = blockIdx.x & 63;
    const int h = (blockIdx.x >> 6) & 3;
    const int b = blockIdx.x >> 8;
    const int tid = threadIdx.x;                  // one query row per thread
    const int tq = bucket * WIN + tid;

    const size_t qoff = ((size_t)(b * TSEQ + tq)) * DIMX + h * DH;
    float qr[DH];
    #pragma unroll
    for (int e = 0; e < DH; ++e) qr[e] = bf2f(qb[qoff + e]);

    float m = -3.4e38f, l = 0.f;
    float acc[DH];
    #pragma unroll
    for (int e = 0; e < DH; ++e) acc[e] = 0.f;

    __shared__ float ks[64][DH];
    __shared__ float vs[64][DH];

    for (int c = 0; c < 4; ++c) {                 // 4 chunks of 64 keys = 256
        __syncthreads();
        for (int idx = tid; idx < 64 * DH; idx += 128) {
            int jj = idx >> 6, e = idx & 63;
            int j = c * 64 + jj;
            int tk = (j < WIN) ? (bucket - 1) * WIN + j : bucket * WIN + (j - WIN);
            bool valid = (j >= WIN) || (bucket > 0);
            float kv = 0.f, vv = 0.f;
            if (valid) {
                size_t o = ((size_t)(b * TSEQ + tk)) * DIMX + h * DH + e;
                kv = bf2f(kb[o]);
                vv = bf2f(vb[o]);
            }
            ks[jj][e] = kv;
            vs[jj][e] = vv;
        }
        __syncthreads();
        for (int jj = 0; jj < 64; ++jj) {
            int j = c * 64 + jj;
            int tk = (j < WIN) ? (bucket - 1) * WIN + j : bucket * WIN + (j - WIN);
            if ((j < WIN && bucket == 0) || (tk > tq)) continue;   // pad | causal
            float s = 0.f;
            #pragma unroll
            for (int e = 0; e < DH; ++e) s += qr[e] * ks[jj][e];
            s *= 0.125f;                                           // DH^-0.5
            float mn = fmaxf(m, s);
            float corr = __expf(m - mn);
            float p = __expf(s - mn);
            l = l * corr + p;
            #pragma unroll
            for (int e = 0; e < DH; ++e) acc[e] = acc[e] * corr + p * vs[jj][e];
            m = mn;
        }
    }
    float inv = 1.0f / l;
    #pragma unroll
    for (int e = 0; e < DH; ++e) attnb[qoff + e] = f2bf(acc[e] * inv);
}

// ---------------- causal linear attention (heads 4..15) ----------------------
__global__ __launch_bounds__(64)
void linear_attn_kernel(const unsigned short* __restrict__ qb,
                        const unsigned short* __restrict__ kb,
                        const unsigned short* __restrict__ vb,
                        unsigned short* __restrict__ attnb) {
    const int hh = blockIdx.x % 12;
    const int b  = blockIdx.x / 12;
    const int h  = LHEADS + hh;
    const int t  = threadIdx.x;                   // 0..63

    __shared__ float qs[BSPOT][DH];
    __shared__ float kp[BSPOT][DH];
    __shared__ float ctx[DH][DH];                 // running sum_n k[n,d]*v[n,e]
    __shared__ float cumk[DH];

    for (int i = t; i < DH * DH; i += 64) ((float*)ctx)[i] = 0.f;
    cumk[t] = 0.f;
    __syncthreads();

    const size_t base = ((size_t)b * TSEQ) * DIMX + h * DH;

    for (int u = 0; u < TSEQ / BSPOT; ++u) {
        const int t0 = u * BSPOT;
        // phase 1: thread = row n. softmax(q) and elu(k)+1
        {
            size_t ro = base + (size_t)(t0 + t) * DIMX;
            float qrow[DH];
            float mx = -3.4e38f;
            #pragma unroll
            for (int d = 0; d < DH; ++d) { qrow[d] = bf2f(qb[ro + d]); mx = fmaxf(mx, qrow[d]); }
            float s = 0.f;
            #pragma unroll
            for (int d = 0; d < DH; ++d) { qrow[d] = __expf(qrow[d] - mx); s += qrow[d]; }
            float inv = 1.0f / s;
            #pragma unroll
            for (int d = 0; d < DH; ++d) qs[t][d] = qrow[d] * inv;
            #pragma unroll
            for (int d = 0; d < DH; ++d) {
                float x = bf2f(kb[ro + d]);
                kp[t][d] = (x > 0.f) ? (x + 1.0f) : __expf(x);
            }
        }
        __syncthreads();
        // phase 2: thread = column e. output uses state of chunks < u (causal shift)
        {
            float cn[DH];
            #pragma unroll
            for (int d = 0; d < DH; ++d) cn[d] = ctx[d][t] * (1.0f / (cumk[d] + 1e-6f));
            for (int n = 0; n < BSPOT; ++n) {
                float s = 0.f;
                #pragma unroll
                for (int d = 0; d < DH; ++d) s += qs[n][d] * cn[d];
                attnb[base + (size_t)(t0 + n) * DIMX + t] = f2bf(s);
            }
        }
        __syncthreads();
        // phase 3: thread = column e. fold chunk u into state
        {
            float vcol[BSPOT];
            #pragma unroll
            for (int n = 0; n < BSPOT; ++n)
                vcol[n] = bf2f(vb[base + (size_t)(t0 + n) * DIMX + t]);
            for (int d = 0; d < DH; ++d) {
                float s = 0.f;
                #pragma unroll
                for (int n = 0; n < BSPOT; ++n) s += kp[n][d] * vcol[n];
                ctx[d][t] += s;
            }
            float cs = 0.f;
            #pragma unroll
            for (int n = 0; n < BSPOT; ++n) cs += kp[n][t];
            cumk[t] += cs;
        }
        __syncthreads();
    }
}

// ---------------- output GEMM: attn @ Wo + bo (WMMA, f32 out) ----------------
__global__ __launch_bounds__(256)
void out_gemm_kernel(const unsigned short* __restrict__ ab,
                     const unsigned short* __restrict__ wot,
                     const float* __restrict__ bo,
                     float* __restrict__ out) {
    __shared__ uint4 sA[2048];                    // 16 x 1024 bf16 = 32KB

    const int wave = threadIdx.x >> 5;
    const int lane = threadIdx.x & 31;
    const int mt = blockIdx.x >> 3;
    const int ng = blockIdx.x & 7;
    const int m0 = mt * 16;
    const int n0 = ng * 128 + wave * 16;
    const int hl  = lane >> 4;
    const int lid = lane & 15;

    stage_A_panel(ab, m0, sA);
    const char* sAc = (const char*)sA;

    v8f c = {};
    const size_t colB = (size_t)(n0 + lid) * DIMX;

    for (int k0 = 0; k0 < DIMX; k0 += 32) {
        __builtin_prefetch(wot + colB + k0 + 32, 0, 0);
        Frag a, f;
        load_A_frag(a, sAc, lid, hl, k0);
        load_B_frag(f, wot, colB, hl, k0);
        c = __builtin_amdgcn_wmma_f32_16x16x32_bf16(false, a.v, false, f.v,
                                                    (short)0, c, false, false);
    }
    #pragma unroll
    for (int g = 0; g < 8; ++g) {
        int row = m0 + g + 8 * hl;
        int col = n0 + lid;
        out[(size_t)row * DIMX + col] = c[g] + bo[col];
    }
}

// ---------------- launcher ----------------
extern "C" void kernel_launch(void* const* d_in, const int* in_sizes, int n_in,
                              void* d_out, int out_size, void* d_ws, size_t ws_size,
                              hipStream_t stream) {
    const float* x  = (const float*)d_in[0];
    const float* Wq = (const float*)d_in[1];
    const float* Wk = (const float*)d_in[2];
    const float* Wv = (const float*)d_in[3];
    const float* Wo = (const float*)d_in[4];
    const float* bo = (const float*)d_in[5];
    float* out = (float*)d_out;

    const size_t NX = (size_t)NB * TSEQ * DIMX;   // 33,554,432
    const size_t NW = (size_t)DIMX * DIMX;        // 1,048,576

    char* w = (char*)d_ws;
    unsigned short* xb  = (unsigned short*)w; w += NX * 2;
    unsigned short* wqt = (unsigned short*)w; w += NW * 2;
    unsigned short* wkt = (unsigned short*)w; w += NW * 2;
    unsigned short* wvt = (unsigned short*)w; w += NW * 2;
    unsigned short* wot = (unsigned short*)w; w += NW * 2;
    unsigned short* qb  = (unsigned short*)w; w += NX * 2;
    unsigned short* kb  = (unsigned short*)w; w += NX * 2;
    unsigned short* vb  = (unsigned short*)w; w += NX * 2;
    unsigned short* ab  = (unsigned short*)w; w += NX * 2;

    cvt_bf16_kernel<<<(unsigned)((NX + 255) / 256), 256, 0, stream>>>(x, xb, (int)NX);
    transpose_w_kernel<<<(unsigned)(NW / 256), 256, 0, stream>>>(Wq, wqt);
    transpose_w_kernel<<<(unsigned)(NW / 256), 256, 0, stream>>>(Wk, wkt);
    transpose_w_kernel<<<(unsigned)(NW / 256), 256, 0, stream>>>(Wv, wvt);
    transpose_w_kernel<<<(unsigned)(NW / 256), 256, 0, stream>>>(Wo, wot);

    // 2048 M-panels x 8 N-groups; 8 waves (16-col tiles) per 256-thread block
    qkv_gemm_kernel<<<16384, 256, 0, stream>>>(xb, wqt, wkt, wvt, qb, kb, vb);

    local_attn_kernel<<<NB * LHEADS * (TSEQ / WIN), 128, 0, stream>>>(qb, kb, vb, ab);
    linear_attn_kernel<<<NB * 12, 64, 0, stream>>>(qb, kb, vb, ab);

    out_gemm_kernel<<<16384, 256, 0, stream>>>(ab, wot, bo, out);
}
